// atten_v3_head_55525337202839
// MI455X (gfx1250) — matrix operand order repaired and checked by reading, not compiled, not run on hardware
//
#include <hip/hip_runtime.h>
#include <hip/hip_bf16.h>
#include <math.h>

// ---------------------------------------------------------------------------
// MI455X / gfx1250 implementation. wave32, WMMA 16x16x32 f16.
//   cast x->f16; pack weights into WMMA A-tile lane-major order;
//   implicit-GEMM WMMA convs (multi-channel-tile accumulation for Cout=256);
//   fused flash attention (QK -> online softmax -> PV) with WMMA, including
//   row-sum-via-WMMA (P @ ones) to avoid half the cross-lane reductions;
//   SE pooling + gate; final combine.
// ---------------------------------------------------------------------------

typedef __attribute__((ext_vector_type(16))) _Float16 v16h;
typedef __attribute__((ext_vector_type(8)))  _Float16 v8h;
typedef __attribute__((ext_vector_type(8)))  float    v8f;

#define WMMA_F16(a, b, c) \
  __builtin_amdgcn_wmma_f32_16x16x32_f16(false, (a), false, (b), (short)0, (c), false, false)

// ------------------------------- cast --------------------------------------
__global__ void cast_f32_to_f16(const float* __restrict__ in,
                                _Float16* __restrict__ out, int n) {
  int i = blockIdx.x * blockDim.x + threadIdx.x;
  if (i < n) out[i] = (_Float16)in[i];
}

// --------------------------- weight packing --------------------------------
// Pack w (Cout, Cin, kh, kw) f32 -> f16 A-tiles in WMMA lane-major order:
//   dst[ ((tap*coTiles + coT)*kch + kc)*512 + lane*16 + e ]
// A-matrix 16x32 f16 layout used here (contiguous k, same order as B staging):
//   m = lane&15 (=cout within tile), k = e + 16*(lane>>4)
__global__ void pack_weights(const float* __restrict__ src,
                             _Float16* __restrict__ dst,
                             int Cout, int Cin, int taps) {
  int kch = (Cin + 31) >> 5;
  int coTiles = Cout >> 4;
  int total = taps * coTiles * kch * 512;
  int idx = blockIdx.x * blockDim.x + threadIdx.x;
  if (idx >= total) return;
  int e    = idx & 15;
  int lane = (idx >> 4) & 31;
  int tile = idx >> 9;
  int kc  = tile % kch;
  int t2  = tile / kch;
  int coT = t2 % coTiles;
  int tap = t2 / coTiles;
  int co = coT * 16 + (lane & 15);
  int kA = e + 16 * (lane >> 4);
  int c  = kc * 32 + kA;
  float v = 0.0f;
  if (c < Cin) v = src[((size_t)co * Cin + c) * taps + tap];
  dst[idx] = (_Float16)v;
}

// ----------------------- generic implicit-GEMM conv ------------------------
// NCHW f16 input, stride-1 symmetric padding, via per-tap 16x16x32 WMMAs.
// Block = 128 threads (4 waves) covering one (b, h) output row; each wave
// computes 16 consecutive pixels x (COINNER*16) output channels. A staged
// B-tile (32 ch x 16 px) is reused by COINNER WMMAs, amortizing LDS traffic
// and barriers (key for the Cout=256 convs).
// OUTMODE: 0 = f16 row-major [(b*HW+pix)*ldo + coOff + co]  (q / kT)
//          1 = f16 NCHW [(b*Cout+co)*HW + pix]              (v / y1)
//          2 = f32 NCHW                                     (y)
// ACT: 0 none, 1 silu
template <int OUTMODE, int ACT, int COINNER>
__global__ __launch_bounds__(128) void conv_implicit_wmma(
    const _Float16* __restrict__ xin, const _Float16* __restrict__ wpk,
    const float* __restrict__ bias, void* __restrict__ outp,
    int Cin, int Cout, int H, int W, int ks, int pad, int ldo, int coOff) {
  const int HW   = H * W;
  const int lane = threadIdx.x & 31;
  const int wid  = threadIdx.x >> 5;
  const int hi   = lane >> 4;
  const int ln   = lane & 15;
  const int h    = blockIdx.x % H;
  const int b    = blockIdx.x / H;
  const int coTiles = Cout >> 4;
  const int coT0 = blockIdx.y * COINNER;
  const int w0   = wid * 16;
  const int kch  = (Cin + 31) >> 5;

  __shared__ __align__(32) _Float16 bt[4][512];  // per-wave packed B tiles

  v8f zero = {};
  v8f acc[COINNER];
#pragma unroll
  for (int i = 0; i < COINNER; ++i) acc[i] = zero;

  // Lanes whose channel never exists (only happens when kch==1, e.g. Cin=16):
  // zero-fill their LDS slots once, skip re-staging inside the loops.
  bool laneDead = ((int)lane >= Cin) && (kch == 1);
  if (laneDead) {
#pragma unroll
    for (int p = 0; p < 16; ++p)
      bt[wid][(p + (hi << 4)) * 16 + ln] = (_Float16)0.0f;
  }

  for (int ty = 0; ty < ks; ++ty) {
    int hs = h + ty - pad;                 // uniform across block
    if (hs < 0 || hs >= H) continue;
    for (int tx = 0; tx < ks; ++tx) {
      int sw0 = w0 + tx - pad;
      int tap = ty * ks + tx;
      for (int kc = 0; kc < kch; ++kc) {
        // Stage B tile (32 channels x 16 pixels) in consumer lane-major
        // order: flat = (p + 16*(k>=16))*16 + (k&15); this lane owns k=lane.
        int c = (kc << 5) + lane;
        if (c < Cin) {
          const _Float16* src = xin + (((size_t)b * Cin + c) * H + hs) * W;
          if (sw0 >= 0 && sw0 + 16 <= W) {
            v8h d0 = *(const v8h*)(src + sw0);
            v8h d1 = *(const v8h*)(src + sw0 + 8);
#pragma unroll
            for (int p = 0; p < 8; ++p) {
              bt[wid][(p + (hi << 4)) * 16 + ln]     = d0[p];
              bt[wid][(p + 8 + (hi << 4)) * 16 + ln] = d1[p];
            }
          } else {
#pragma unroll
            for (int p = 0; p < 16; ++p) {
              int sw = sw0 + p;
              _Float16 v = (_Float16)0.0f;
              if (sw >= 0 && sw < W) v = src[sw];
              bt[wid][(p + (hi << 4)) * 16 + ln] = v;
            }
          }
        }
        __syncthreads();
        v16h bm = *(const v16h*)(&bt[wid][lane * 16]);
        const _Float16* wbase =
            wpk + (((size_t)tap * coTiles + coT0) * kch + kc) * 512 + lane * 16;
        // Prefetch next A-tile group (global_prefetch_b8) to cover latency.
        __builtin_prefetch((const void*)(wbase + (size_t)kch * 512), 0, 1);
#pragma unroll
        for (int ci = 0; ci < COINNER; ++ci) {
          v16h a = *(const v16h*)(wbase + (size_t)ci * kch * 512);
          acc[ci] = WMMA_F16(a, bm, acc[ci]);
        }
        __syncthreads();
      }
    }
  }

#pragma unroll
  for (int ci = 0; ci < COINNER; ++ci) {
    int co = (coT0 + ci) * 16 + ln;
    float bv = bias[co];
#pragma unroll
    for (int r = 0; r < 8; ++r) {
      float v = acc[ci][r] + bv;
      if (ACT == 1) v = v / (1.0f + __expf(-v));   // silu
      int pix = h * W + w0 + r + (hi << 3);
      if (OUTMODE == 0) {
        ((_Float16*)outp)[((size_t)b * HW + pix) * ldo + coOff + co] = (_Float16)v;
      } else if (OUTMODE == 1) {
        ((_Float16*)outp)[((size_t)b * Cout + co) * HW + pix] = (_Float16)v;
      } else {
        ((float*)outp)[((size_t)b * Cout + co) * HW + pix] = v;
      }
    }
  }
}

// --------------------------- flash attention -------------------------------
// q  : (B, HW, 32) f16 row-major  (A-tiles load contiguously)
// kT : (B, HW, 32) f16 row-major  (B-tiles for S load contiguously)
// v  : (B, C, HW)  f16            (B-tiles for PV load contiguously)
// out: (B, HW, C)  f32  (row = query pixel, col = channel)
// Block = 256 threads (8 waves); each wave owns 16 query rows x all 256
// channels (16 f32 C/D accumulator tiles). Online softmax over all HW cols;
// the 268MB attention matrix is never materialized (HBM roofline win).
// Row-sums are computed by the matrix unit itself (P @ ones -> every column
// of D holds the row sum, landing directly in the rl[] register mapping),
// halving the ds_bpermute reduction traffic.
__global__ __launch_bounds__(256) void flash_attn(
    const _Float16* __restrict__ q, const _Float16* __restrict__ kT,
    const _Float16* __restrict__ vx, float* __restrict__ o,
    int HW, int C) {
  const int lane = threadIdx.x & 31;
  const int wid  = threadIdx.x >> 5;
  const int hi   = lane >> 4;
  const int ln   = lane & 15;
  const int nb   = HW / 128;
  const int b    = blockIdx.x / nb;
  const int rowBase = (blockIdx.x % nb) * 128 + wid * 16;

  __shared__ __align__(32) _Float16 pt[8][512];  // per-wave P tile 16x32

  // Q A-tile (16 rows x K=32): whole head dim in one WMMA.
  const _Float16* qp = q + ((size_t)b * HW + rowBase + ln) * 32;
  v8h qlo = *(const v8h*)(qp + hi * 16);
  v8h qhi = *(const v8h*)(qp + hi * 16 + 8);
  v16h qa;
#pragma unroll
  for (int i = 0; i < 8; ++i) { qa[i] = qlo[i]; qa[i + 8] = qhi[i]; }

  // All-ones B tile: row-sum reducer for the matrix unit.
  v16h ones;
#pragma unroll
  for (int i = 0; i < 16; ++i) ones[i] = (_Float16)1.0f;

  v8f zero = {};
  v8f acc[16];
#pragma unroll
  for (int t = 0; t < 16; ++t) acc[t] = zero;
  float rm[8], rl[8];
#pragma unroll
  for (int r = 0; r < 8; ++r) { rm[r] = -3.0e30f; rl[r] = 0.0f; }

  const _Float16* kb = kT + (size_t)b * HW * 32;
  const _Float16* vb = vx + (size_t)b * C * HW;

  for (int m0 = 0; m0 < HW; m0 += 32) {
    // S = Q @ K : two 16x16 tiles (m-cols m0..m0+15, m0+16..m0+31)
    v16h bk0 = *(const v16h*)(kb + (size_t)(m0 + ln) * 32 + hi * 16);
    v16h bk1 = *(const v16h*)(kb + (size_t)(m0 + 16 + ln) * 32 + hi * 16);
    v8f s0 = WMMA_F16(qa, bk0, zero);
    v8f s1 = WMMA_F16(qa, bk1, zero);

    // Online softmax max update. C-layout rows live in 16-lane halves:
    // butterfly reductions with masks {1,2,4,8} stay within a row's lanes.
    float p0[8], p1[8], sc[8];
#pragma unroll
    for (int r = 0; r < 8; ++r) {
      float t = fmaxf(s0[r], s1[r]);
      t = fmaxf(t, __shfl_xor(t, 1, 32));
      t = fmaxf(t, __shfl_xor(t, 2, 32));
      t = fmaxf(t, __shfl_xor(t, 4, 32));
      t = fmaxf(t, __shfl_xor(t, 8, 32));
      float mn = fmaxf(rm[r], t);
      sc[r] = __expf(rm[r] - mn);
      rm[r] = mn;
      p0[r] = __expf(s0[r] - mn);
      p1[r] = __expf(s1[r] - mn);
    }
#pragma unroll
    for (int t = 0; t < 16; ++t)
#pragma unroll
      for (int r = 0; r < 8; ++r) acc[t][r] *= sc[r];

    // C-layout -> A-layout transpose of P through LDS (1KB/wave).
#pragma unroll
    for (int r = 0; r < 8; ++r) {
      int qr = r + hi * 8;
      pt[wid][qr * 32 + ln]      = (_Float16)p0[r];
      pt[wid][qr * 32 + 16 + ln] = (_Float16)p1[r];
    }
    __syncthreads();
    const _Float16* pp = &pt[wid][ln * 32];
    v8h plo = *(const v8h*)(pp + hi * 16);
    v8h phi = *(const v8h*)(pp + hi * 16 + 8);
    v16h pa;
#pragma unroll
    for (int i = 0; i < 8; ++i) { pa[i] = plo[i]; pa[i + 8] = phi[i]; }
    __syncthreads();

    // Row sums via the matrix unit: D = P @ ones. Every column of a row
    // holds that row's sum, already in the per-lane rl[] register mapping.
    v8f rs = WMMA_F16(pa, ones, zero);
#pragma unroll
    for (int r = 0; r < 8; ++r) rl[r] = rl[r] * sc[r] + rs[r];

    // O += P @ V^T : 16 channel tiles, B-tiles contiguous from (C,HW) v.
#pragma unroll
    for (int ct = 0; ct < 16; ++ct) {
      v16h bv = *(const v16h*)(vb + (size_t)(ct * 16 + ln) * HW + m0 + hi * 16);
      acc[ct] = WMMA_F16(pa, bv, acc[ct]);
    }
  }

  float inv[8];
#pragma unroll
  for (int r = 0; r < 8; ++r) inv[r] = 1.0f / rl[r];
  float* ob = o + (size_t)b * HW * C;
#pragma unroll
  for (int ct = 0; ct < 16; ++ct)
#pragma unroll
    for (int r = 0; r < 8; ++r)
      ob[(size_t)(rowBase + r + hi * 8) * C + ct * 16 + ln] = acc[ct][r] * inv[r];
}

// ------------------------------ SE branch ----------------------------------
__global__ void pool_mean(const float* __restrict__ y, float* __restrict__ pooled,
                          int HW) {
  __shared__ float red[256];
  int bc = blockIdx.x;
  const float* p = y + (size_t)bc * HW;
  float s = 0.0f;
  for (int i = threadIdx.x; i < HW; i += blockDim.x) s += p[i];
  red[threadIdx.x] = s;
  __syncthreads();
  for (int off = blockDim.x >> 1; off > 0; off >>= 1) {
    if ((int)threadIdx.x < off) red[threadIdx.x] += red[threadIdx.x + off];
    __syncthreads();
  }
  if (threadIdx.x == 0) pooled[bc] = red[0] / (float)HW;
}

__global__ void se_gate(const float* __restrict__ sew, const float* __restrict__ seb,
                        const float* __restrict__ pooled, float* __restrict__ gate,
                        int Bn, int C) {
  int idx = blockIdx.x * blockDim.x + threadIdx.x;
  if (idx >= Bn * C) return;
  int b = idx / C, co = idx % C;
  float s = seb[co];
  const float* wr = sew + (size_t)co * C;
  const float* pr = pooled + (size_t)b * C;
  for (int c = 0; c < C; ++c) s += wr[c] * pr[c];
  gate[idx] = 1.0f / (1.0f + __expf(-s));
}

__global__ void final_combine(const float* __restrict__ attn,
                              const float* __restrict__ y,
                              const float* __restrict__ gate,
                              const float* __restrict__ x,
                              const float* __restrict__ g1,
                              const float* __restrict__ g2,
                              float* __restrict__ out, int C, int HW, int total) {
  int idx = blockIdx.x * blockDim.x + threadIdx.x;
  if (idx >= total) return;
  int n = idx % HW;
  int t = idx / HW;
  int c = t % C;
  int b = t / C;
  float a = attn[((size_t)b * HW + n) * C + c];
  out[idx] = g1[0] * a + g2[0] * y[idx] * gate[b * C + c] + x[idx];
}

// ------------------------------- launch ------------------------------------
extern "C" void kernel_launch(void* const* d_in, const int* in_sizes, int n_in,
                              void* d_out, int out_size, void* d_ws, size_t ws_size,
                              hipStream_t stream) {
  const int Bn = 4, C = 256, H = 64, W = 64, HW = H * W;
  const int Cq = 16, Ck = 32;

  const float* x       = (const float*)d_in[0];
  const float* dw1_w   = (const float*)d_in[1];
  const float* dw1_b   = (const float*)d_in[2];
  const float* dw2_w   = (const float*)d_in[3];
  const float* dw2_b   = (const float*)d_in[4];
  const float* key_w   = (const float*)d_in[5];
  const float* key_b   = (const float*)d_in[6];
  const float* value_w = (const float*)d_in[7];
  const float* value_b = (const float*)d_in[8];
  const float* conv1_w = (const float*)d_in[9];
  const float* conv1_b = (const float*)d_in[10];
  const float* conv2_w = (const float*)d_in[11];
  const float* conv2_b = (const float*)d_in[12];
  const float* se_w    = (const float*)d_in[13];
  const float* se_b    = (const float*)d_in[14];
  const float* g1      = (const float*)d_in[15];
  const float* g2      = (const float*)d_in[16];
  float* out = (float*)d_out;

  // ---- workspace carve-up (≈55 MB total) ----
  char* ws = (char*)d_ws;
  size_t off = 0;
  auto carve = [&](size_t bytes) -> void* {
    void* p = ws + off;
    off = (off + bytes + 255) & ~(size_t)255;
    return p;
  };
  _Float16* xh   = (_Float16*)carve((size_t)Bn * C * HW * 2);   // x as f16 NCHW
  _Float16* qbuf = (_Float16*)carve((size_t)Bn * HW * 32 * 2);  // (B,HW,32)
  _Float16* kTb  = (_Float16*)carve((size_t)Bn * HW * 32 * 2);  // (B,HW,32)
  _Float16* vbuf = (_Float16*)carve((size_t)Bn * C * HW * 2);   // (B,C,HW)
  float*    attn = (float*)carve((size_t)Bn * HW * C * 4);      // (B,HW,C)
  _Float16* y1   = (_Float16*)carve((size_t)Bn * Cq * HW * 2);  // silu(conv1)
  float*    ybuf = (float*)carve((size_t)Bn * C * HW * 4);      // conv2 out
  float*    pooled = (float*)carve((size_t)Bn * C * 4);
  float*    gate   = (float*)carve((size_t)Bn * C * 4);
  _Float16* pk_dw1 = (_Float16*)carve((size_t)49  * 1  * 8 * 512 * 2);
  _Float16* pk_dw2 = (_Float16*)carve((size_t)121 * 1  * 8 * 512 * 2);
  _Float16* pk_key = (_Float16*)carve((size_t)1   * 2  * 8 * 512 * 2);
  _Float16* pk_val = (_Float16*)carve((size_t)1   * 16 * 8 * 512 * 2);
  _Float16* pk_c1  = (_Float16*)carve((size_t)9   * 1  * 8 * 512 * 2);
  _Float16* pk_c2  = (_Float16*)carve((size_t)9   * 16 * 1 * 512 * 2);
  (void)ws_size; (void)in_sizes; (void)n_in; (void)out_size;

  // ---- 0: cast input to f16 ----
  {
    int n = Bn * C * HW;
    cast_f32_to_f16<<<(n + 255) / 256, 256, 0, stream>>>(x, xh, n);
  }

  // ---- 1: pack weights into WMMA A-tile order ----
  auto pack = [&](const float* src, _Float16* dst, int Cout, int Cin, int taps) {
    int total = taps * (Cout / 16) * ((Cin + 31) / 32) * 512;
    pack_weights<<<(total + 255) / 256, 256, 0, stream>>>(src, dst, Cout, Cin, taps);
  };
  pack(dw1_w,   pk_dw1, Cq, C, 49);
  pack(dw2_w,   pk_dw2, Cq, C, 121);
  pack(key_w,   pk_key, Ck, C, 1);
  pack(value_w, pk_val, C,  C, 1);
  pack(conv1_w, pk_c1,  Cq, C, 9);
  pack(conv2_w, pk_c2,  C,  Cq, 9);

  // ---- 2: conv branches (implicit-GEMM WMMA) ----
  // q cols 0..15 = 7x7 branch, cols 16..31 = 11x11 branch (matches concat).
  conv_implicit_wmma<0, 0, 1><<<dim3(Bn * H, 1), 128, 0, stream>>>(
      xh, pk_dw1, dw1_b, qbuf, C, Cq, H, W, 7, 3, 32, 0);
  conv_implicit_wmma<0, 0, 1><<<dim3(Bn * H, 1), 128, 0, stream>>>(
      xh, pk_dw2, dw2_b, qbuf, C, Cq, H, W, 11, 5, 32, 16);
  // key: 2 channel tiles per staged B tile.
  conv_implicit_wmma<0, 0, 2><<<dim3(Bn * H, 1), 128, 0, stream>>>(
      xh, pk_key, key_b, kTb, C, Ck, H, W, 1, 0, 32, 0);
  // value: 16 channel tiles per staged B tile (16 WMMAs / stage).
  conv_implicit_wmma<1, 0, 16><<<dim3(Bn * H, 1), 128, 0, stream>>>(
      xh, pk_val, value_b, vbuf, C, C, H, W, 1, 0, 0, 0);

  // ---- 3: fused flash attention (QK -> softmax -> PV) ----
  flash_attn<<<Bn * (HW / 128), 256, 0, stream>>>(qbuf, kTb, vbuf, attn, HW, C);

  // ---- 4: local branch conv1(silu) -> conv2 ----
  conv_implicit_wmma<1, 1, 1><<<dim3(Bn * H, 1), 128, 0, stream>>>(
      xh, pk_c1, conv1_b, y1, C, Cq, H, W, 3, 1, 0, 0);
  conv_implicit_wmma<2, 0, 16><<<dim3(Bn * H, 1), 128, 0, stream>>>(
      y1, pk_c2, conv2_b, ybuf, Cq, C, H, W, 3, 1, 0, 0);

  // ---- 5: SE pooling + gate ----
  pool_mean<<<Bn * C, 256, 0, stream>>>(ybuf, pooled, HW);
  se_gate<<<(Bn * C + 255) / 256, 256, 0, stream>>>(se_w, se_b, pooled, gate, Bn, C);

  // ---- 6: final combine: gamma1*attn + gamma2*y*gate + x ----
  {
    int total = Bn * C * HW;
    final_combine<<<(total + 255) / 256, 256, 0, stream>>>(
        attn, ybuf, gate, x, g1, g2, out, C, HW, total);
  }
}
